// DRNNBert_21328807592514
// MI455X (gfx1250) — matrix operand order repaired and checked by reading, not compiled, not run on hardware
//
#include <hip/hip_runtime.h>

// ---------------------------------------------------------------------------
// Dual-layer tanh RNN (B=32, L=512, E=768, H=1024) + two heads (O=128) on
// MI455X (gfx1250). All GEMMs use V_WMMA_F32_16X16X32_BF16 (wave32).
//   - weights pre-packed on device into WMMA B-fragment order (bf16)
//   - activations row-major bf16 -> A-fragment = 2 aligned b128 loads/lane
//   - recurrence: persistent kernel, 16 blocks x 8 waves = 128 waves
//     (exactly one 16x16 tile of the 32x1024 h_t per wave), software
//     grid barrier between the 512 sequential steps
// ---------------------------------------------------------------------------

typedef __bf16 v16bf __attribute__((ext_vector_type(16)));
typedef float  v8f   __attribute__((ext_vector_type(8)));

#define B_   32
#define L_   512
#define E_   768
#define H_   1024
#define O_   128
#define TOK_ (B_ * L_)

static __device__ __forceinline__ unsigned short f2bf(float f) {
  unsigned u = __float_as_uint(f);
  u += 0x7fffu + ((u >> 16) & 1u);       // round-to-nearest-even
  return (unsigned short)(u >> 16);
}

static __device__ __forceinline__ v8f wmma_bf16(v16bf a, v16bf b, v8f c) {
  // (neg_a, A, neg_b, B, c_mod, C, reuse_a, reuse_b)
  return __builtin_amdgcn_wmma_f32_16x16x32_bf16(false, a, false, b,
                                                 (short)0, c, false, false);
}

union FragU { v16bf v; uint4 q[2]; };

// A-fragment (16x32 bf16, row-major source, K contiguous).
// Lane l<16: row M=l, K = kbase+[0..15];
// lane>=16: row M=l-16, K = kbase+[16..31].
static __device__ __forceinline__ v16bf load_afrag(
    const unsigned short* rowbase, int kbase, int lane) {
  const unsigned short* p = rowbase + kbase + ((lane >> 4) << 4);
  FragU u;
  u.q[0] = *reinterpret_cast<const uint4*>(p);
  u.q[1] = *reinterpret_cast<const uint4*>(p + 8);
  return u.v;
}

// B-fragment: pre-packed, lane's 16 halves contiguous (32B per lane).
static __device__ __forceinline__ v16bf load_bfrag(
    const unsigned short* fragbase, int lane) {
  const unsigned short* p = fragbase + lane * 16;
  FragU u;
  u.q[0] = *reinterpret_cast<const uint4*>(p);
  u.q[1] = *reinterpret_cast<const uint4*>(p + 8);
  return u.v;
}

// ------------------------- prep kernels ------------------------------------

__global__ void k_cvt_bf16(const float* __restrict__ in,
                           unsigned short* __restrict__ out, int n) {
  int i = blockIdx.x * blockDim.x + threadIdx.x;
  if (i < n) out[i] = f2bf(in[i]);
}

// Pack W (O rows x K cols, f32) used as B = W^T (K x O) into fragment order:
// frag = nt*KT + kk ; within frag: lane l (0..31), elem e (0..15):
//   n = 16*nt + (l&15), k = 32*kk + 16*(l>=16) + e, value = W[n*K + k]
__global__ void k_pack(const float* __restrict__ W,
                       unsigned short* __restrict__ out, int K, int total) {
  int i = blockIdx.x * blockDim.x + threadIdx.x;
  if (i >= total) return;
  int frag = i >> 9;
  int rr   = i & 511;
  int lane = rr >> 4;
  int e    = rr & 15;
  int KT   = K >> 5;
  int nt   = frag / KT;
  int kk   = frag - nt * KT;
  int n    = (nt << 4) + (lane & 15);
  int k    = (kk << 5) + ((lane >> 4) << 4) + e;
  out[i] = f2bf(W[(size_t)n * K + k]);
}

__global__ void k_bsum(const float* __restrict__ a, const float* __restrict__ b,
                       float* __restrict__ o, int n) {
  int i = blockIdx.x * blockDim.x + threadIdx.x;
  if (i < n) o[i] = a[i] + b[i];
}

// ------------------------- generic WMMA GEMM -------------------------------
// D[M x N] (f32, row-major) = A[M x K] (bf16 row-major) * Bpacked + bias[N]
// One 16x16 tile per wave; grid = (M/16)*(N/16)/8 blocks of 256 threads.
__global__ void k_gemm(const unsigned short* __restrict__ A,
                       const unsigned short* __restrict__ Bp,
                       const float* __restrict__ bias,
                       float* __restrict__ D, int N, int K) {
  const int lane = threadIdx.x & 31;
  const int wid  = blockIdx.x * (blockDim.x >> 5) + (threadIdx.x >> 5);
  const int NT = N >> 4, KT = K >> 5;
  const int mt = wid / NT, nt = wid - mt * NT;
  const int col   = (nt << 4) + (lane & 15);
  const int arow  = (mt << 4) + (lane & 15);
  const unsigned short* arowp = A + (size_t)arow * K;
  const unsigned short* bfrag = Bp + (size_t)nt * KT * 512;

  const float bv = bias ? bias[col] : 0.0f;
  v8f acc;
#pragma unroll
  for (int i = 0; i < 8; ++i) acc[i] = bv;

  for (int kk = 0; kk < KT; ++kk) {
    __builtin_prefetch(bfrag + 512, 0, 0);   // global_prefetch_b8 next frag
    v16bf a = load_afrag(arowp, kk << 5, lane);
    v16bf b = load_bfrag(bfrag, lane);
    acc = wmma_bf16(a, b, acc);
    bfrag += 512;
  }

  const int rbase = (mt << 4) + ((lane >> 4) << 3);
#pragma unroll
  for (int v = 0; v < 8; ++v)
    D[(size_t)(rbase + v) * N + col] = acc[v];
}

// ------------------------- persistent recurrence ---------------------------
// h_t = tanh(xp_t + h_{t-1} W_hh^T). Launch as exactly 16 blocks x 256 thr
// (128 waves == 128 output tiles of the 32x1024 hidden state). Grid barrier
// between timesteps via monotone counter in workspace (reset per launch).
__global__ void k_rnn(const float* __restrict__ xp,          // [TOK, H] f32
                      const unsigned short* __restrict__ Whhp,
                      unsigned short* __restrict__ r,        // [TOK, H] bf16
                      unsigned short* __restrict__ hbuf,     // 2 x [B_, H] bf16
                      unsigned* __restrict__ cnt) {
  const int lane = threadIdx.x & 31;
  const int wid  = blockIdx.x * 8 + (threadIdx.x >> 5);  // 0..127
  const int mt = wid >> 6, nt = wid & 63;
  const int col   = (nt << 4) + (lane & 15);
  const int arow  = (mt << 4) + (lane & 15);
  const int rbase = (mt << 4) + ((lane >> 4) << 3);
  const unsigned short* bbase = Whhp + (size_t)nt * 32 * 512;  // KT = 32

  unsigned phase = 0;
  for (int t = 0; t < L_; ++t) {
    v8f acc;
#pragma unroll
    for (int v = 0; v < 8; ++v) {
      int brow = rbase + v;  // batch row 0..31
      acc[v] = xp[((size_t)brow * L_ + t) * H_ + col];
    }

    if (t > 0) {  // h_0 == 0: skip the matmul on the first step
      const unsigned short* hprev = hbuf + (size_t)((t & 1) ^ 1) * (B_ * H_);
      const unsigned short* arowp = hprev + (size_t)arow * H_;
      const unsigned short* bfrag = bbase;
      for (int kk = 0; kk < 32; ++kk) {
        __builtin_prefetch(bfrag + 512, 0, 0);
        v16bf a = load_afrag(arowp, kk << 5, lane);
        v16bf b = load_bfrag(bfrag, lane);
        acc = wmma_bf16(a, b, acc);
        bfrag += 512;
      }
    }

    unsigned short* hcur = hbuf + (size_t)(t & 1) * (B_ * H_);
#pragma unroll
    for (int v = 0; v < 8; ++v) {
      unsigned short hb = f2bf(tanhf(acc[v]));
      int brow = rbase + v;
      hcur[(size_t)brow * H_ + col] = hb;
      r[((size_t)brow * L_ + t) * H_ + col] = hb;
    }

    // grid-wide barrier (16 blocks, all co-resident)
    __syncthreads();
    if (threadIdx.x == 0) {
      __threadfence();
      atomicAdd(cnt, 1u);
      ++phase;
      while (__hip_atomic_load(cnt, __ATOMIC_ACQUIRE,
                               __HIP_MEMORY_SCOPE_AGENT) < phase * 16u) {
        __builtin_amdgcn_s_sleep(1);
      }
    }
    __syncthreads();
  }
}

// ------------------------- launch ------------------------------------------

extern "C" void kernel_launch(void* const* d_in, const int* in_sizes, int n_in,
                              void* d_out, int out_size, void* d_ws,
                              size_t ws_size, hipStream_t stream) {
  (void)in_sizes; (void)n_in; (void)out_size; (void)ws_size;
  const float* x    = (const float*)d_in[0];
  const float* Wih1 = (const float*)d_in[1];
  const float* Whh1 = (const float*)d_in[2];
  const float* bih1 = (const float*)d_in[3];
  const float* bhh1 = (const float*)d_in[4];
  const float* Wih2 = (const float*)d_in[5];
  const float* Whh2 = (const float*)d_in[6];
  const float* bih2 = (const float*)d_in[7];
  const float* bhh2 = (const float*)d_in[8];
  const float* W1   = (const float*)d_in[9];
  const float* b1   = (const float*)d_in[10];
  const float* W2   = (const float*)d_in[11];
  const float* b2   = (const float*)d_in[12];

  char* ws = (char*)d_ws;
  size_t off = 0;
  auto carve = [&](size_t bytes) -> char* {
    char* p = ws + off;
    off = (off + bytes + 255) & ~(size_t)255;
    return p;
  };
  unsigned*       cnt   = (unsigned*)      carve(256);
  unsigned short* xbf   = (unsigned short*)carve((size_t)TOK_ * E_ * 2);
  unsigned short* r1    = (unsigned short*)carve((size_t)TOK_ * H_ * 2);
  unsigned short* r2    = (unsigned short*)carve((size_t)TOK_ * H_ * 2);
  float*          xp    = (float*)         carve((size_t)TOK_ * H_ * 4);
  unsigned short* hbuf  = (unsigned short*)carve((size_t)2 * B_ * H_ * 2);
  unsigned short* Wih1p = (unsigned short*)carve((size_t)E_ * H_ * 2);
  unsigned short* Whh1p = (unsigned short*)carve((size_t)H_ * H_ * 2);
  unsigned short* Wih2p = (unsigned short*)carve((size_t)H_ * H_ * 2);
  unsigned short* Whh2p = (unsigned short*)carve((size_t)H_ * H_ * 2);
  unsigned short* W1p   = (unsigned short*)carve((size_t)H_ * O_ * 2);
  unsigned short* W2p   = (unsigned short*)carve((size_t)H_ * O_ * 2);
  float*          bs1   = (float*)         carve(H_ * 4);
  float*          bs2   = (float*)         carve(H_ * 4);

  hipMemsetAsync(cnt, 0, 256, stream);  // barrier counter (deterministic)

  int n;
  n = TOK_ * E_; k_cvt_bf16<<<(n + 255) / 256, 256, 0, stream>>>(x, xbf, n);
  n = E_ * H_;   k_pack<<<(n + 255) / 256, 256, 0, stream>>>(Wih1, Wih1p, E_, n);
  n = H_ * H_;   k_pack<<<(n + 255) / 256, 256, 0, stream>>>(Whh1, Whh1p, H_, n);
  n = H_ * H_;   k_pack<<<(n + 255) / 256, 256, 0, stream>>>(Wih2, Wih2p, H_, n);
  n = H_ * H_;   k_pack<<<(n + 255) / 256, 256, 0, stream>>>(Whh2, Whh2p, H_, n);
  n = H_ * O_;   k_pack<<<(n + 255) / 256, 256, 0, stream>>>(W1, W1p, H_, n);
  n = H_ * O_;   k_pack<<<(n + 255) / 256, 256, 0, stream>>>(W2, W2p, H_, n);
  k_bsum<<<(H_ + 255) / 256, 256, 0, stream>>>(bih1, bhh1, bs1, H_);
  k_bsum<<<(H_ + 255) / 256, 256, 0, stream>>>(bih2, bhh2, bs2, H_);

  // layer 1: xp1 = x * W_ih1^T + (b_ih1 + b_hh1), then recurrence -> r1
  {
    int waves = (TOK_ / 16) * (H_ / 16);
    k_gemm<<<waves / 8, 256, 0, stream>>>(xbf, Wih1p, bs1, xp, H_, E_);
  }
  k_rnn<<<16, 256, 0, stream>>>(xp, Whh1p, r1, hbuf, cnt);
  hipMemsetAsync(cnt, 0, 256, stream);  // reset barrier for layer 2

  // layer 2: xp2 = r1 * W_ih2^T + (b_ih2 + b_hh2), then recurrence -> r2
  {
    int waves = (TOK_ / 16) * (H_ / 16);
    k_gemm<<<waves / 8, 256, 0, stream>>>(r1, Wih2p, bs2, xp, H_, H_);
  }
  k_rnn<<<16, 256, 0, stream>>>(xp, Whh2p, r2, hbuf, cnt);

  // heads: y = r1 W1^T + b1 ; z = r2 W2^T + b2  (concatenated in d_out)
  {
    int waves = (TOK_ / 16) * (O_ / 16);
    float* y = (float*)d_out;
    float* z = y + (size_t)TOK_ * O_;
    k_gemm<<<waves / 8, 256, 0, stream>>>(r1, W1p, b1, y, O_, H_);
    k_gemm<<<waves / 8, 256, 0, stream>>>(r2, W2p, b2, z, O_, H_);
  }
}
